// WinogradConv2d_13176959664495
// MI455X (gfx1250) — matrix operand order repaired and checked by reading, not compiled, not run on hardware
//
#include <hip/hip_runtime.h>
#include <hip/hip_bf16.h>

typedef __attribute__((ext_vector_type(16))) _Float16 v16h;
typedef __attribute__((ext_vector_type(8)))  float    v8f;

union FragU { uint4 u[2]; v16h h; };

static __device__ __forceinline__ unsigned pack2h(float a, float b) {
    _Float16 ha = (_Float16)a, hb = (_Float16)b;
    unsigned short ua, ub;
    __builtin_memcpy(&ua, &ha, 2);
    __builtin_memcpy(&ub, &hb, 2);
    return (unsigned)ua | ((unsigned)ub << 16);
}

// ---------------------------------------------------------------------------
// Filter transform: U[f][c][4][4] = G K G^T, stored pre-swizzled as WMMA
// B-fragments (f16 pairs) at Uws[((pos*2+fh)*32 + lane)*8 + v].
// B layout (32x16 f16): lanes 0-15 hold K(=c)={0-7,16-23}, lanes 16-31 hold c={8-15,24-31};
// VGPR v halfword h -> c = 16*(v/4) + 8*(lane/16) + 2*(v%4) + h. Column N = filter = lane%16 (+16*fh).
// ---------------------------------------------------------------------------
__global__ void winograd_filter_transform(const float* __restrict__ Kw,
                                          unsigned* __restrict__ Uws) {
    int tid = blockIdx.x * blockDim.x + threadIdx.x;  // 0..511
    if (tid >= 512) return;
    int f  = tid >> 4;         // 0..31
    int c0 = (tid & 15) * 2;   // even channel of the pair

    float Uv[2][16];
    #pragma unroll
    for (int s = 0; s < 2; ++s) {
        int c = c0 + s;
        const float* kp = Kw + (f * 32 + c) * 9;
        float k0[3] = {kp[0], kp[1], kp[2]};
        float k1[3] = {kp[3], kp[4], kp[5]};
        float k2[3] = {kp[6], kp[7], kp[8]};
        float t[4][3];
        #pragma unroll
        for (int j = 0; j < 3; ++j) {
            t[0][j] = k0[j];
            t[1][j] = 0.5f * (k0[j] + k1[j] + k2[j]);
            t[2][j] = 0.5f * (k0[j] - k1[j] + k2[j]);
            t[3][j] = k2[j];
        }
        #pragma unroll
        for (int i = 0; i < 4; ++i) {
            Uv[s][i*4+0] = t[i][0];
            Uv[s][i*4+1] = 0.5f * (t[i][0] + t[i][1] + t[i][2]);
            Uv[s][i*4+2] = 0.5f * (t[i][0] - t[i][1] + t[i][2]);
            Uv[s][i*4+3] = t[i][2];
        }
    }
    int fh   = f >> 4;
    int lane = ((c0 >> 3) & 1) * 16 + (f & 15);
    int v    = (c0 >> 4) * 4 + ((c0 & 7) >> 1);
    #pragma unroll
    for (int pos = 0; pos < 16; ++pos)
        Uws[(((pos * 2 + fh) * 32 + lane) << 3) + v] = pack2h(Uv[0][pos], Uv[1][pos]);
}

// ---------------------------------------------------------------------------
// Main kernel: one wave handles (batch b, tile-row p, 16 consecutive q-tiles),
// all 32 channels, all 32 filters (two 16-filter WMMA halves).
// ---------------------------------------------------------------------------
__global__ void __launch_bounds__(256)
winograd_main(const float* __restrict__ X,
              const unsigned* __restrict__ Uws,
              float* __restrict__ Y) {
    extern __shared__ unsigned lds[];
    const int lane = threadIdx.x & 31;
    const int wave = threadIdx.x >> 5;
    unsigned* wlds = lds + wave * 4096;   // 16 KB (4096 dwords) per wave

    const int wg  = blockIdx.x * 8 + wave;   // 0..16383
    const int b   = wg >> 10;                // batch 0..15
    const int grp = wg & 1023;
    const int p   = grp >> 3;                // tile row 0..127
    const int qb  = (grp & 7) << 4;          // base tile col (multiple of 16)

    // ---- Phase 1: input transform V = BT d BT^T, scatter f16 pairs into LDS
    // A layout (16x32 f16): frag-lane L<16 -> tile=L, channels {0-7,16-23};
    // L>=16 -> tile=L-16, channels {8-15,24-31}. dword idx = pos*256 + L*8 + v.
    #pragma unroll
    for (int k = 0; k < 8; ++k) {
        int item  = k * 32 + lane;
        int tile  = item & 15;
        int c0    = (item >> 4) * 2;   // even channel of pair, 0..30
        float Vv[2][16];
        #pragma unroll
        for (int s = 0; s < 2; ++s) {
            int c = c0 + s;
            const float* xp = X + ((size_t)(b * 32 + c) * 258 + (size_t)(2 * p)) * 258
                                + 2 * (qb + tile);
            float d[4][4];
            #pragma unroll
            for (int r = 0; r < 4; ++r) {
                const float* row = xp + (size_t)r * 258;
                float2 a0 = *(const float2*)(row);
                float2 a1 = *(const float2*)(row + 2);
                d[r][0] = a0.x; d[r][1] = a0.y; d[r][2] = a1.x; d[r][3] = a1.y;
            }
            float t[4][4];
            #pragma unroll
            for (int kk = 0; kk < 4; ++kk) {
                t[0][kk] = d[0][kk] - d[2][kk];
                t[1][kk] = d[1][kk] + d[2][kk];
                t[2][kk] = d[2][kk] - d[1][kk];
                t[3][kk] = d[1][kk] - d[3][kk];
            }
            #pragma unroll
            for (int i = 0; i < 4; ++i) {
                Vv[s][i*4+0] = t[i][0] - t[i][2];
                Vv[s][i*4+1] = t[i][1] + t[i][2];
                Vv[s][i*4+2] = t[i][2] - t[i][1];
                Vv[s][i*4+3] = t[i][1] - t[i][3];
            }
        }
        int g        = c0 >> 3;                              // channel group 0..3
        int fragLane = (g & 1) * 16 + tile;
        int v        = (g >> 1) * 4 + ((c0 & 7) >> 1);
        unsigned base = fragLane * 8 + v;
        #pragma unroll
        for (int pos = 0; pos < 16; ++pos)
            wlds[pos * 256 + base] = pack2h(Vv[0][pos], Vv[1][pos]);
    }
    __syncthreads();

    // ---- Phase 2: 16 WMMAs per filter-half, then inverse transform A^T M A.
    #pragma unroll
    for (int fh = 0; fh < 2; ++fh) {
        v8f acc[16];
        #pragma unroll
        for (int pos = 0; pos < 16; ++pos) {
            FragU a, bm;
            const uint4* ap = (const uint4*)(wlds + pos * 256 + lane * 8);
            a.u[0] = ap[0]; a.u[1] = ap[1];
            const uint4* bp = (const uint4*)(Uws + (((pos * 2 + fh) * 32 + lane) << 3));
            bm.u[0] = bp[0]; bm.u[1] = bp[1];
            v8f cz = {};
            acc[pos] = __builtin_amdgcn_wmma_f32_16x16x32_f16(
                false, a.h, false, bm.h, (short)0, cz, false, false);
        }
        // C/D layout: lane n holds filter n%16, tiles {8*(n/16)+r : r=0..7} in VGPR r.
        int f     = fh * 16 + (lane & 15);
        int tbase = (lane >> 4) * 8;
        #pragma unroll
        for (int r = 0; r < 8; ++r) {
            int tile = tbase + r;
            float m[16];
            #pragma unroll
            for (int pos = 0; pos < 16; ++pos) m[pos] = acc[pos][r];
            float tmp[2][4];
            #pragma unroll
            for (int l = 0; l < 4; ++l) {
                tmp[0][l] = m[0*4+l] + m[1*4+l] + m[2*4+l];
                tmp[1][l] = m[1*4+l] - m[2*4+l] - m[3*4+l];
            }
            float y00 = tmp[0][0] + tmp[0][1] + tmp[0][2];
            float y01 = tmp[0][1] - tmp[0][2] - tmp[0][3];
            float y10 = tmp[1][0] + tmp[1][1] + tmp[1][2];
            float y11 = tmp[1][1] - tmp[1][2] - tmp[1][3];
            int q = qb + tile;
            float* yp = Y + ((size_t)(b * 32 + f) * 256 + (size_t)(2 * p)) * 256 + 2 * q;
            *(float2*)yp         = make_float2(y00, y01);
            *(float2*)(yp + 256) = make_float2(y10, y11);
        }
    }
}

extern "C" void kernel_launch(void* const* d_in, const int* in_sizes, int n_in,
                              void* d_out, int out_size, void* d_ws, size_t ws_size,
                              hipStream_t stream) {
    const float* X  = (const float*)d_in[0];   // (16, 32, 258, 258) fp32
    const float* Kw = (const float*)d_in[1];   // (32, 32, 3, 3)     fp32
    float* Y        = (float*)d_out;           // (16, 32, 256, 256) fp32
    unsigned* Uws   = (unsigned*)d_ws;         // 32 KB of pre-swizzled f16 U

    winograd_filter_transform<<<2, 256, 0, stream>>>(Kw, Uws);
    // 16 batches * 1024 tile-groups = 16384 waves / 8 waves-per-block = 2048 blocks
    winograd_main<<<2048, 256, 131072, stream>>>(X, Uws, Y);
}
